// ScaledDotProductAttention_65609920414166
// MI455X (gfx1250) — matrix operand order repaired and checked by reading, not compiled, not run on hardware
//
#include <hip/hip_runtime.h>
#include <hip/hip_bf16.h>

typedef __attribute__((ext_vector_type(16))) __bf16       bf16x16;
typedef __attribute__((ext_vector_type(8)))  float        f32x8;
typedef __attribute__((ext_vector_type(4)))  float        f32x4;
typedef __attribute__((ext_vector_type(4)))  unsigned int u32x4;
typedef __attribute__((ext_vector_type(8)))  int          i32x8;
typedef __attribute__((ext_vector_type(4)))  int          i32x4;

#define B_  4
#define S_  4096
#define D_  64
#define KT  64      // keys per tile
#define BQ  64      // queries per block (4 waves x 16)
#define NTHREADS 128
#define LSTR 72     // padded LDS row stride in bf16 elems (144B: conflict-free b128)

union Frag  { u32x4 q[2]; bf16x16 v; };
union Pack8 { __bf16 h[8]; u32x4 v; unsigned int u[4]; };

__device__ __forceinline__ f32x8 wmma_bf16(bf16x16 a, bf16x16 b, f32x8 c) {
  return __builtin_amdgcn_wmma_f32_16x16x32_bf16(false, a, false, b, (short)0, c,
                                                 false, false);
}

// xor-16 lane exchange within wave32 (group-of-32 swizzle: xor=0x10, and=0x1f)
__device__ __forceinline__ float swz16f(float x) {
  return __int_as_float(__builtin_amdgcn_ds_swizzle(__float_as_int(x), 0x401f));
}

__device__ __forceinline__ bf16x16 load_frag_lds(const __bf16* row, int e0, int e1) {
  Frag f;
  f.q[0] = *(const u32x4*)(row + e0);
  f.q[1] = *(const u32x4*)(row + e1);
  return f.v;
}

__device__ __forceinline__ bf16x16 load_q_frag(const float* qrow, int d0) {
  Frag f; Pack8 a, b;
#pragma unroll
  for (int i = 0; i < 8; ++i) a.h[i] = (__bf16)(qrow[d0 + i] * 0.125f);
#pragma unroll
  for (int i = 0; i < 8; ++i) b.h[i] = (__bf16)(qrow[d0 + 8 + i] * 0.125f);
  f.q[0] = a.v; f.q[1] = b.v;
  return f.v;
}

// ---------------- TDM descriptor construction (ISA 08_async_tensor §8) -------
__device__ __forceinline__ u32x4 tdm_g0(unsigned lds_addr, unsigned long long ga) {
  u32x4 g;
  g[0] = 1u;                                   // count=1 (valid), user mode
  g[1] = lds_addr;                             // LDS byte address
  g[2] = (unsigned)ga;                         // global_addr[31:0]
  g[3] = (unsigned)((ga >> 32) & 0x01ffffffull) | (2u << 30);  // addr[56:32]|type=2
  return g;
}
// 2-D tile, 2-byte elements, LDS row padding 128B-data + 16B-pad (stride 144B)
__device__ __forceinline__ i32x8 tdm_g1(unsigned tile0, unsigned tile1,
                                        unsigned td0, unsigned td1,
                                        unsigned long long stride0) {
  i32x8 g;
  g[0] = (int)((1u << 16)     // data_size = 2 bytes
             | (1u << 20)     // pad_enable
             | (4u << 22)     // pad_interval: 32 dwords (=128B) before pad
             | (3u << 25));   // pad_amount: 4 dwords (=16B)
  g[1] = (int)((td0 & 0xffffu) << 16);                   // tensor_dim0[15:0]
  g[2] = (int)((td0 >> 16) | ((td1 & 0xffffu) << 16));   // dim0 hi | dim1 lo
  g[3] = (int)((td1 >> 16) | (tile0 << 16));             // dim1 hi | tile_dim0
  g[4] = (int)tile1;                                     // tile_dim1 (tile_dim2=0)
  g[5] = (int)(unsigned)stride0;                         // tensor_dim0_stride lo
  g[6] = (int)((unsigned)(stride0 >> 32) & 0xffffu);     // stride0 hi | stride1=0
  g[7] = 0;
  return g;
}

__device__ __forceinline__ unsigned lds_off(const void* p) {
  // generic LDS pointer = {aperture_hi32, lds_offset_lo32} (ISA §10.2)
  return (unsigned)(unsigned long long)(size_t)p;
}

// ---------------- shared per-wave flash-attention tile --------------------
struct WaveCtx {
  bf16x16 bq0, bq1;
  f32x8   oacc[4];
  float   mrun, lrun;
  int     ln, hh, qme;
};

__device__ __forceinline__ void attn_tile(WaveCtx& w, int k0, bool causal,
                                          const __bf16* Kl, const __bf16* Vl,
                                          const float* mpen) {
  const f32x8 zf = {0.f, 0.f, 0.f, 0.f, 0.f, 0.f, 0.f, 0.f};
  const int eA0 = 16 * w.hh,      eA1 = 8 + 16 * w.hh;
  const int eB0 = 32 + 16 * w.hh, eB1 = 40 + 16 * w.hh;
  const float* mpl = mpen + 8 * w.hh;

  // S^T = K_tile x Q^T : four 16-key blocks, head dim split 2x32
  f32x8 s[4];
#pragma unroll
  for (int kb = 0; kb < 4; ++kb) {
    const __bf16* kr = Kl + (size_t)(16 * kb + w.ln) * LSTR;
    s[kb] = wmma_bf16(load_frag_lds(kr, eA0, eA1), w.bq0, zf);
    s[kb] = wmma_bf16(load_frag_lds(kr, eB0, eB1), w.bq1, s[kb]);
  }

  // additive penalties: key padding always; causal only on the diagonal tile
  float pv[4][8];
#pragma unroll
  for (int kb = 0; kb < 4; ++kb) {
    const f32x4 ma = *(const f32x4*)(mpl + 16 * kb);
    const f32x4 mb = *(const f32x4*)(mpl + 16 * kb + 4);
#pragma unroll
    for (int r = 0; r < 8; ++r) {
      float pen = (r < 4) ? ma[r] : mb[r - 4];
      if (causal)
        pen += ((k0 + 16 * kb + 8 * w.hh + r) > w.qme) ? -10000.0f : 0.0f;
      pv[kb][r] = s[kb][r] + pen;
    }
  }

  // online softmax: in-lane reduce over 32 keys + one xor16 exchange
  float tmax = -INFINITY;
#pragma unroll
  for (int kb = 0; kb < 4; ++kb)
#pragma unroll
    for (int r = 0; r < 8; ++r) tmax = fmaxf(tmax, pv[kb][r]);
  tmax = fmaxf(tmax, swz16f(tmax));
  const float mnew  = fmaxf(w.mrun, tmax);
  const float alpha = __expf(w.mrun - mnew);
  w.mrun = mnew;

  float psum = 0.0f;
#pragma unroll
  for (int kb = 0; kb < 4; ++kb)
#pragma unroll
    for (int r = 0; r < 8; ++r) {
      pv[kb][r] = __expf(pv[kb][r] - mnew);
      psum += pv[kb][r];
    }
  psum += swz16f(psum);
  w.lrun = w.lrun * alpha + psum;

#pragma unroll
  for (int c = 0; c < 4; ++c)
#pragma unroll
    for (int r = 0; r < 8; ++r) w.oacc[c][r] *= alpha;

  // two P^T B-fragments: bf16 pack + half-swap between lane pairs (l, l^16)
  bf16x16 pf[2];
#pragma unroll
  for (int f = 0; f < 2; ++f) {
    Pack8 own0, own1;
#pragma unroll
    for (int r = 0; r < 8; ++r) {
      own0.h[r] = (__bf16)pv[2 * f][r];
      own1.h[r] = (__bf16)pv[2 * f + 1][r];
    }
    Frag pp; Pack8 lo, hi;
#pragma unroll
    for (int i = 0; i < 4; ++i) {
      const unsigned int send = w.hh ? own0.u[i] : own1.u[i];
      const unsigned int recv =
          (unsigned int)__builtin_amdgcn_ds_swizzle((int)send, 0x401f);
      lo.u[i] = w.hh ? recv      : own0.u[i];
      hi.u[i] = w.hh ? own1.u[i] : recv;
    }
    pp.q[0] = lo.v; pp.q[1] = hi.v;
    pf[f] = pp.v;
  }

  // O^T += V^T x P^T
#pragma unroll
  for (int c = 0; c < 4; ++c) {
    const __bf16* vr = Vl + (size_t)(16 * c + w.ln) * LSTR;
    w.oacc[c] = wmma_bf16(load_frag_lds(vr, eA0, eA1), pf[0], w.oacc[c]);
    w.oacc[c] = wmma_bf16(load_frag_lds(vr, eB0, eB1), pf[1], w.oacc[c]);
  }
}

// =================== pre-pass: f32 -> bf16, V transposed =====================
__global__ __launch_bounds__(256) void convert_kv_kernel(
    const float* __restrict__ K, const float* __restrict__ V,
    __bf16* __restrict__ Kbf, __bf16* __restrict__ Vtbf) {
  __shared__ __bf16 vt[64][72];
  const int b   = blockIdx.x / (S_ / 64);
  const int s0  = (blockIdx.x % (S_ / 64)) * 64;
  const int tid = threadIdx.x;
  const int sl  = tid >> 2;
  const int c0  = (tid & 3) * 16;

  const float* ks = K + ((size_t)b * S_ + s0 + sl) * D_ + c0;
  __bf16*      kd = Kbf + ((size_t)b * S_ + s0 + sl) * D_ + c0;
#pragma unroll
  for (int i = 0; i < 16; ++i) kd[i] = (__bf16)ks[i];

  const float* vs = V + ((size_t)b * S_ + s0 + sl) * D_ + c0;
#pragma unroll
  for (int i = 0; i < 16; ++i) vt[c0 + i][sl] = (__bf16)vs[i];
  __syncthreads();

  __bf16* vd = Vtbf + ((size_t)b * D_ + sl) * S_ + s0 + c0;
#pragma unroll
  for (int i = 0; i < 16; ++i) vd[i] = vt[sl][c0 + i];
}

// =================== main kernel: TDM double-buffered staging ================
__global__ __launch_bounds__(NTHREADS) void
fa_fwd_tdm_kernel(const float* __restrict__ Q, const int* __restrict__ mask,
                  const __bf16* __restrict__ Kbf, const __bf16* __restrict__ Vtbf,
                  float* __restrict__ O) {
  __shared__ __align__(16) __bf16 Kbuf[2][KT][LSTR];
  __shared__ __align__(16) __bf16 Vbuf[2][D_][LSTR];
  __shared__ __align__(16) float  maskpen[2][KT];

  const int tid  = threadIdx.x;
  const int bIdx = blockIdx.x / (S_ / BQ);
  const int qb   = (blockIdx.x % (S_ / BQ)) * BQ;
  const int wave = tid >> 5;
  const int lane = tid & 31;

  WaveCtx w;
  w.ln  = lane & 15;
  w.hh  = lane >> 4;
  w.qme = qb + wave * 16 + w.ln;
  w.mrun = -INFINITY;
  w.lrun = 0.0f;
  const f32x8 zf = {0.f, 0.f, 0.f, 0.f, 0.f, 0.f, 0.f, 0.f};
#pragma unroll
  for (int c = 0; c < 4; ++c) w.oacc[c] = zf;

  const float* qrow = Q + ((size_t)bIdx * S_ + w.qme) * D_;
  w.bq0 = load_q_frag(qrow, w.hh * 16);
  w.bq1 = load_q_frag(qrow, 32 + w.hh * 16);

  const int* Mb = mask + (size_t)bIdx * S_;

  const unsigned lk[2] = { lds_off(&Kbuf[0][0][0]), lds_off(&Kbuf[1][0][0]) };
  const unsigned lv[2] = { lds_off(&Vbuf[0][0][0]), lds_off(&Vbuf[1][0][0]) };
  const i32x4 z4 = {0, 0, 0, 0};
  const i32x8 z8 = {0, 0, 0, 0, 0, 0, 0, 0};

  auto issue_tile = [&](int k0, int buf) __attribute__((always_inline)) {
    // K tile: 64 keys x 64 dims, contiguous rows of 64 elems
    const unsigned long long gaK =
        (unsigned long long)(size_t)(Kbf + ((size_t)bIdx * S_ + k0) * D_);
    __builtin_amdgcn_tensor_load_to_lds(
        tdm_g0(lk[buf], gaK), tdm_g1(64, 64, 64, 64, 64ull), z4, z4, z8, 0);
    // V^T tile: 64 dims x 64 keys, row stride S elems
    const unsigned long long gaV =
        (unsigned long long)(size_t)(Vtbf + (size_t)bIdx * D_ * S_ + k0);
    __builtin_amdgcn_tensor_load_to_lds(
        tdm_g0(lv[buf], gaV), tdm_g1(64, 64, 64, 64, (unsigned long long)S_),
        z4, z4, z8, 0);
  };

  const int ntiles = qb / KT + 1;   // interior tiles + one diagonal tile

  // prologue: kick off tile 0 into buffer 0
  if (wave == 0) issue_tile(0, 0);
  if (tid < KT) maskpen[0][tid] = Mb[tid] ? 0.0f : -10000.0f;
  __syncthreads();

  // interior tiles: causal logic compiled out, DMA for j+1 overlaps compute of j
  for (int j = 0; j < ntiles - 1; ++j) {
    const int cur = j & 1;
    if (wave == 0) issue_tile((j + 1) * KT, cur ^ 1);
    if (tid < KT)
      maskpen[cur ^ 1][tid] = Mb[(j + 1) * KT + tid] ? 0.0f : -10000.0f;
    if (wave == 0) __builtin_amdgcn_s_wait_tensorcnt(2);  // tile j complete
    __syncthreads();                                      // broadcast to waves

    attn_tile(w, j * KT, /*causal=*/false,
              &Kbuf[cur][0][0], &Vbuf[cur][0][0], &maskpen[cur][0]);
    __syncthreads();                // done reading buf cur before reuse
  }

  // peeled diagonal tile: constant causal=true
  {
    const int cur = (ntiles - 1) & 1;
    if (wave == 0) __builtin_amdgcn_s_wait_tensorcnt(0);
    __syncthreads();
    attn_tile(w, (ntiles - 1) * KT, /*causal=*/true,
              &Kbuf[cur][0][0], &Vbuf[cur][0][0], &maskpen[cur][0]);
  }

  const float invl = 1.0f / w.lrun;
  float* orow = O + (size_t)(bIdx * S_ + w.qme) * D_;
#pragma unroll
  for (int c = 0; c < 4; ++c)
#pragma unroll
    for (int r = 0; r < 8; ++r)
      orow[16 * c + r + 8 * w.hh] = w.oacc[c][r] * invl;
}

// =================== fallback (no workspace): VALU staging ===================
__global__ __launch_bounds__(NTHREADS) void
fa_fwd_fallback_kernel(const float* __restrict__ Q, const float* __restrict__ K,
                       const float* __restrict__ V, const int* __restrict__ mask,
                       float* __restrict__ O) {
  __shared__ __align__(16) __bf16 Klds[KT][LSTR];
  __shared__ __align__(16) __bf16 Vtlds[D_][LSTR];
  __shared__ __align__(16) float  maskpen[KT];

  const int tid  = threadIdx.x;
  const int bIdx = blockIdx.x / (S_ / BQ);
  const int qb   = (blockIdx.x % (S_ / BQ)) * BQ;
  const int wave = tid >> 5;
  const int lane = tid & 31;

  WaveCtx w;
  w.ln  = lane & 15;
  w.hh  = lane >> 4;
  w.qme = qb + wave * 16 + w.ln;
  w.mrun = -INFINITY;
  w.lrun = 0.0f;
  const f32x8 zf = {0.f, 0.f, 0.f, 0.f, 0.f, 0.f, 0.f, 0.f};
#pragma unroll
  for (int c = 0; c < 4; ++c) w.oacc[c] = zf;

  const float* Qb = Q + (size_t)bIdx * S_ * D_;
  const float* Kb = K + (size_t)bIdx * S_ * D_;
  const float* Vb = V + (size_t)bIdx * S_ * D_;
  const int*   Mb = mask + (size_t)bIdx * S_;

  const float* qrow = Qb + (size_t)w.qme * D_;
  w.bq0 = load_q_frag(qrow, w.hh * 16);
  w.bq1 = load_q_frag(qrow, 32 + w.hh * 16);

  auto stage = [&](int k0, bool pref) __attribute__((always_inline)) {
    {
      const int row = tid >> 1, c0 = (tid & 1) * 32;
      const float* ks = Kb + (size_t)(k0 + row) * D_ + c0;
#pragma unroll
      for (int g = 0; g < 4; ++g) {
        Pack8 p;
#pragma unroll
        for (int i = 0; i < 8; ++i) p.h[i] = (__bf16)ks[8 * g + i];
        *(u32x4*)&Klds[row][c0 + 8 * g] = p.v;
      }
    }
    {
      const int dd = tid & 63, kp = (tid >> 6) * 32;
      const float* vs = Vb + (size_t)(k0 + kp) * D_ + dd;
#pragma unroll
      for (int g = 0; g < 4; ++g) {
        Pack8 p;
#pragma unroll
        for (int i = 0; i < 8; ++i) p.h[i] = (__bf16)vs[(size_t)(8 * g + i) * D_];
        *(u32x4*)&Vtlds[dd][kp + 8 * g] = p.v;
      }
    }
    if (tid < KT) {
      maskpen[tid] = Mb[k0 + tid] ? 0.0f : -10000.0f;
      if (pref) {
        __builtin_prefetch(Kb + (size_t)(k0 + KT + tid) * D_, 0, 1);
        __builtin_prefetch(Vb + (size_t)(k0 + KT + tid) * D_, 0, 1);
      }
    }
  };

  const int nInterior = qb / KT;
  for (int j = 0; j < nInterior; ++j) {
    stage(j * KT, true);
    __syncthreads();
    attn_tile(w, j * KT, false, &Klds[0][0], &Vtlds[0][0], &maskpen[0]);
    __syncthreads();
  }
  stage(nInterior * KT, false);
  __syncthreads();
  attn_tile(w, nInterior * KT, true, &Klds[0][0], &Vtlds[0][0], &maskpen[0]);

  const float invl = 1.0f / w.lrun;
  float* orow = O + (size_t)(bIdx * S_ + w.qme) * D_;
#pragma unroll
  for (int c = 0; c < 4; ++c)
#pragma unroll
    for (int r = 0; r < 8; ++r)
      orow[16 * c + r + 8 * w.hh] = w.oacc[c][r] * invl;
}

extern "C" void kernel_launch(void* const* d_in, const int* in_sizes, int n_in,
                              void* d_out, int out_size, void* d_ws, size_t ws_size,
                              hipStream_t stream) {
  (void)in_sizes; (void)n_in; (void)out_size;
  const float* Q    = (const float*)d_in[0];
  const float* K    = (const float*)d_in[1];
  const float* V    = (const float*)d_in[2];
  const int*   mask = (const int*)d_in[3];
  float* O = (float*)d_out;

  dim3 grid(B_ * (S_ / BQ));
  dim3 block(NTHREADS);
  const size_t need = (size_t)B_ * S_ * D_ * sizeof(__bf16) * 2;  // Kbf + Vtbf
  if (d_ws != nullptr && ws_size >= need) {
    __bf16* Kbf  = (__bf16*)d_ws;
    __bf16* Vtbf = Kbf + (size_t)B_ * S_ * D_;
    hipLaunchKernelGGL(convert_kv_kernel, dim3(B_ * (S_ / 64)), dim3(256), 0,
                       stream, K, V, Kbf, Vtbf);
    hipLaunchKernelGGL(fa_fwd_tdm_kernel, grid, block, 0, stream,
                       Q, mask, Kbf, Vtbf, O);
  } else {
    hipLaunchKernelGGL(fa_fwd_fallback_kernel, grid, block, 0, stream,
                       Q, K, V, mask, O);
  }
}
